// VectorQuantizer_4329327034577
// MI455X (gfx1250) — matrix operand order repaired and checked by reading, not compiled, not run on hardware
//
#include <hip/hip_runtime.h>
#include <hip/hip_bf16.h>
#include <stdint.h>

typedef __attribute__((ext_vector_type(16))) __bf16 v16bf;
typedef __attribute__((ext_vector_type(8)))  float  v8f;
typedef __attribute__((ext_vector_type(4)))  unsigned int v4u;
typedef __attribute__((ext_vector_type(8)))  int v8i;
typedef __attribute__((ext_vector_type(4)))  int v4i;

#define NUM_E   512
#define EDIM    64
#define ROWS    128
#define Q_ELEMS 16777216
#define CBBLOB  67584   // 2048 B norms + 65536 B B-fragments, contiguous

// d_ws layout:
//   [0,      2048)  : codebook squared norms, 512 f32
//   [2048,  67584)  : B fragments, bf16, 2 chunks * 32 tiles * 32 lanes * 16 slots
//   [67584, 75776)  : per-block loss partial sums, 2048 f32

__device__ __forceinline__ unsigned short f2bf(float f) {
  union { __bf16 h; unsigned short u; } cv;
  cv.h = (__bf16)f;
  return cv.u;
}

// ---------------- Kernel 0: codebook prep ----------------
__global__ void vq_prep(const float* __restrict__ cb,
                        float* __restrict__ norms,
                        unsigned short* __restrict__ Bfrag) {
  int e = blockIdx.x * blockDim.x + threadIdx.x;
  if (e >= NUM_E) return;
  const float* row = cb + e * EDIM;
  const int ct = e >> 4;     // column tile
  const int nl = e & 15;     // N within tile == lane&15
  float s = 0.f;
  for (int c = 0; c < EDIM; ++c) {
    float v = row[c];
    s += v * v;
    int q    = c >> 5;       // K-chunk (0: c 0..31, 1: c 32..63)
    int kl   = c & 31;       // K within chunk
    int hi   = kl >> 4;      // lane half holds K 16..31
    int slot = kl & 15;      // slot within lane (B layout: K = slot + 16*hi)
    int lane = nl + hi * 16;
    int flat = (((q * 32 + ct) * 32) + lane) * 16 + slot;
    Bfrag[flat] = f2bf(v);
  }
  norms[e] = s;
}

// ---------------- Kernel 1: distances (WMMA) + argmin + quantize + loss ----------------
__global__ __launch_bounds__(256)
void vq_main(const float* __restrict__ in, const float* __restrict__ cb,
             const unsigned char* __restrict__ cbblob,   // norms + B fragments, 67584 B
             float* __restrict__ outQ, float* __restrict__ outIdx,
             float* __restrict__ blockSums) {
  __shared__ __align__(16) unsigned char  lds_blob[CBBLOB];  // [0,2048) norms, [2048,..) B frags
  __shared__ __align__(16) unsigned short lds_x[ROWS * 72];  // bf16 bits, padded rows
  __shared__ int   lds_idx[ROWS];
  __shared__ float lds_ws[8];

  const int t   = threadIdx.x;
  const int n0  = blockIdx.x * ROWS;
  const int b   = n0 >> 15;          // n = b*32768 + (d*1024 + h*32 + w)
  const int rem = n0 & 32767;        // 128 rows stay within one (b); contiguous per channel
  const size_t gbase = (size_t)b * 2097152u + (size_t)rem;

  const int wv   = t >> 5;
  const int lane = t & 31;
  const int hi   = lane >> 4;
  const int nl   = lane & 15;

  // ---- Stage norms + B fragments with the Tensor Data Mover (wave 0 issues) ----
  // 1-D copy: 8448 x 8-byte elements, global cbblob -> LDS blob.
  if (wv == 0) {
    unsigned long long ga = (unsigned long long)(uintptr_t)cbblob;
    unsigned int lds_addr = (unsigned int)(uintptr_t)&lds_blob[0]; // addr[31:0] == LDS address
    v4u g0 = { 1u,                                   // count=1, user mode
               lds_addr,
               (unsigned int)ga,                     // global_addr[31:0]  (D# bits 95:64)
               (unsigned int)((ga >> 32) & 0x1FFFFFFu) | (2u << 30) }; // addr[56:32] | type=2
    v8i g1 = { (int)(3u << 16),                      // data_size = 3 (8 bytes)
               (int)(0x2100u << 16),                 // tensor_dim0 = 8448 (low 16 -> bits 63:48)
               0,                                    // tensor_dim0 hi=0, tensor_dim1=0
               (int)(0x2100u << 16),                 // tile_dim0 = 8448 (bits 127:112)
               0,                                    // tile_dim1 = 0 (unused), tile_dim2 = 0
               8448,                                 // tensor_dim0_stride low 32
               0, 0 };
    v4i gz4 = { 0, 0, 0, 0 };
    v8i gz8 = { 0, 0, 0, 0, 0, 0, 0, 0 };
    __builtin_amdgcn_tensor_load_to_lds(g0, g1, gz4, gz4, gz8, 0);
  }

  // ---- Stage input tile: 128x64 fp32 -> bf16 LDS (coalesced float4 per channel line) ----
  #pragma unroll
  for (int k = 0; k < 8; ++k) {
    int u  = t + k * 256;            // 2048 quads
    int c  = u >> 5;
    int i4 = (u & 31) * 4;
    const float4 x4 = *(const float4*)(in + gbase + (size_t)c * 32768u + i4);
    lds_x[(i4 + 0) * 72 + c] = f2bf(x4.x);
    lds_x[(i4 + 1) * 72 + c] = f2bf(x4.y);
    lds_x[(i4 + 2) * 72 + c] = f2bf(x4.z);
    lds_x[(i4 + 3) * 72 + c] = f2bf(x4.w);
  }
  if (wv == 0) __builtin_amdgcn_s_wait_tensorcnt(0);  // TDM done before barrier
  __syncthreads();

  const float* lds_norm = (const float*)lds_blob;
  const uint4* lds_B    = (const uint4*)(lds_blob + 2048);
  const int irow = wv * 16 + nl;     // A-matrix row for this lane (M = lane&15)

  union U16 { uint4 q[2]; v16bf v; };
  // A fragments (loaded once, reused for all 32 column tiles).
  // A layout: slots 0-15 -> K = hi*16 + 0..15
  U16 ua0, ua1;
  {
    const int c0 = hi * 16;
    ua0.q[0] = *(const uint4*)&lds_x[irow * 72 + c0];
    ua0.q[1] = *(const uint4*)&lds_x[irow * 72 + c0 + 8];
    ua1.q[0] = *(const uint4*)&lds_x[irow * 72 + 32 + c0];
    ua1.q[1] = *(const uint4*)&lds_x[irow * 72 + 32 + c0 + 8];
  }
  const v16bf a0 = ua0.v, a1 = ua1.v;

  float best[8];
  int   bidx[8];
  #pragma unroll
  for (int r = 0; r < 8; ++r) { best[r] = 3.4e38f; bidx[r] = 0; }

  #pragma unroll 2
  for (int ct = 0; ct < 32; ++ct) {
    U16 ub0, ub1;
    const int f0 = (ct * 32 + lane) * 2;        // chunk 0
    const int f1 = ((32 + ct) * 32 + lane) * 2; // chunk 1
    ub0.q[0] = lds_B[f0];  ub0.q[1] = lds_B[f0 + 1];
    ub1.q[0] = lds_B[f1];  ub1.q[1] = lds_B[f1 + 1];
    v8f d = {0.f, 0.f, 0.f, 0.f, 0.f, 0.f, 0.f, 0.f};
    d = __builtin_amdgcn_wmma_f32_16x16x32_bf16(false, a0, false, ub0.v, (short)0, d, false, false);
    d = __builtin_amdgcn_wmma_f32_16x16x32_bf16(false, a1, false, ub1.v, (short)0, d, false, false);
    const int   n  = ct * 16 + nl;     // this lane's codebook column
    const float en = lds_norm[n];
    #pragma unroll
    for (int r = 0; r < 8; ++r) {      // D: lane holds rows M = r + 8*hi for column n
      float s = __builtin_fmaf(-2.f, d[r], en);  // ||e||^2 - 2<x,e>  (argmin-equivalent)
      if (s < best[r]) { best[r] = s; bidx[r] = n; }
    }
  }

  // argmin across the 16 lanes of each half (tie -> lowest index, matching jnp.argmin)
  #pragma unroll
  for (int r = 0; r < 8; ++r) {
    #pragma unroll
    for (int m = 1; m < 16; m <<= 1) {
      float ob = __shfl_xor(best[r], m, 32);
      int   oi = __shfl_xor(bidx[r], m, 32);
      if (ob < best[r] || (ob == best[r] && oi < bidx[r])) { best[r] = ob; bidx[r] = oi; }
    }
  }
  if (nl == 0) {
    const int rbase = wv * 16 + hi * 8;
    #pragma unroll
    for (int r = 0; r < 8; ++r) {
      lds_idx[rbase + r] = bidx[r];
      outIdx[n0 + rbase + r] = (float)bidx[r];
    }
  }
  __syncthreads();

  // quantized output (fp32 codebook gather) + loss partials from original fp32 inputs
  float lsum = 0.f;
  #pragma unroll
  for (int k = 0; k < 8; ++k) {
    int u  = t + k * 256;
    int c  = u >> 5;
    int i4 = (u & 31) * 4;
    const size_t ga = gbase + (size_t)c * 32768u + i4;
    const float4 x4 = *(const float4*)(in + ga);
    float q0 = cb[lds_idx[i4 + 0] * 64 + c];
    float q1 = cb[lds_idx[i4 + 1] * 64 + c];
    float q2 = cb[lds_idx[i4 + 2] * 64 + c];
    float q3 = cb[lds_idx[i4 + 3] * 64 + c];
    *(float4*)(outQ + ga) = make_float4(q0, q1, q2, q3);
    float d0 = q0 - x4.x, d1 = q1 - x4.y, d2 = q2 - x4.z, d3 = q3 - x4.w;
    lsum += d0 * d0 + d1 * d1 + d2 * d2 + d3 * d3;
  }
  #pragma unroll
  for (int m = 1; m < 32; m <<= 1) lsum += __shfl_xor(lsum, m, 32);
  if (lane == 0) lds_ws[wv] = lsum;
  __syncthreads();
  if (t == 0) {
    float s = 0.f;
    #pragma unroll
    for (int w = 0; w < 8; ++w) s += lds_ws[w];
    blockSums[blockIdx.x] = s;   // deterministic fixed-order reduction
  }
}

// ---------------- Kernel 2: final deterministic loss reduction ----------------
__global__ void vq_finish(const float* __restrict__ blockSums,
                          float* __restrict__ outLoss) {
  __shared__ float sh[256];
  float s = 0.f;
  for (int k = 0; k < 8; ++k) s += blockSums[threadIdx.x + k * 256];
  sh[threadIdx.x] = s;
  __syncthreads();
  if (threadIdx.x == 0) {
    float tot = 0.f;
    for (int i = 0; i < 256; ++i) tot += sh[i];
    outLoss[0] = 1.25f * tot / 16777216.0f;   // (1 + commitment_cost) * MSE
  }
}

extern "C" void kernel_launch(void* const* d_in, const int* in_sizes, int n_in,
                              void* d_out, int out_size, void* d_ws, size_t ws_size,
                              hipStream_t stream) {
  (void)in_sizes; (void)n_in; (void)out_size; (void)ws_size;
  const float* in = (const float*)d_in[0];   // [8,64,32,32,32] f32
  const float* cb = (const float*)d_in[1];   // [512,64] f32
  float* outF = (float*)d_out;
  char*  ws   = (char*)d_ws;
  float*          norms = (float*)ws;
  unsigned short* Bfrag = (unsigned short*)(ws + 2048);
  float*          bsums = (float*)(ws + CBBLOB);

  vq_prep<<<2, 256, 0, stream>>>(cb, norms, Bfrag);
  vq_main<<<2048, 256, 0, stream>>>(in, cb, (const unsigned char*)ws,
                                    outF, outF + (Q_ELEMS + 1), bsums);
  vq_finish<<<1, 256, 0, stream>>>(bsums, outF + Q_ELEMS);
}
